// CerebroOriginal_79182017069528
// MI455X (gfx1250) — 8-bit matrix operand order repaired and checked by reading, compiled, not run on hardware
//
#include <hip/hip_runtime.h>
#include <hip/hip_bf16.h>
#include <stdint.h>


typedef int v8i __attribute__((ext_vector_type(8)));

namespace cfg {
constexpr int B  = 256, C = 14, T = 8, E = 180, H = 5, NB = 6, NC = 4;
constexpr int DH = E / H;
constexpr int S  = C * T;
constexpr int MLP = 4 * E;
constexpr int M  = B * S;
constexpr int KP_E   = 192;
constexpr int KP_MLP = 768;
constexpr int NP_E   = 192;
constexpr int NP_MLP = 768;
}

__device__ __forceinline__ float cell_scale(const unsigned* c) {
    return fmaxf(__uint_as_float(*c) * (1.0f / 127.0f), 1e-8f);
}
__device__ __forceinline__ float fakeq(float x, float s) {
    float r = rintf(x / s);
    r = fminf(fmaxf(r, -128.0f), 127.0f);
    return r * s;
}

__global__ void k_reset(unsigned* cells, int n) {
    int i = threadIdx.x;
    if (i < n) cells[i] = 0u;
}

__global__ void k_absmax(const float* __restrict__ x, long n, unsigned* __restrict__ amax) {
    __shared__ float red[256];
    float m = 0.0f;
    for (long i = blockIdx.x * (long)blockDim.x + threadIdx.x; i < n;
         i += (long)gridDim.x * blockDim.x)
        m = fmaxf(m, fabsf(x[i]));
    red[threadIdx.x] = m; __syncthreads();
    for (int s = 128; s > 0; s >>= 1) {
        if (threadIdx.x < s) red[threadIdx.x] = fmaxf(red[threadIdx.x], red[threadIdx.x + s]);
        __syncthreads();
    }
    if (threadIdx.x == 0) atomicMax(amax, __float_as_uint(red[0]));
}

__global__ void k_ln(const float* __restrict__ x, const float* __restrict__ g,
                     const float* __restrict__ b, float* __restrict__ out,
                     unsigned* __restrict__ amax) {
    constexpr int E = cfg::E;
    __shared__ float red[256];
    int row = blockIdx.x, tid = threadIdx.x;
    float v = (tid < E) ? x[(size_t)row * E + tid] : 0.0f;
    red[tid] = v; __syncthreads();
    for (int s = 128; s > 0; s >>= 1) { if (tid < s) red[tid] += red[tid + s]; __syncthreads(); }
    float mean = red[0] / E; __syncthreads();
    float d = (tid < E) ? (v - mean) : 0.0f;
    red[tid] = d * d; __syncthreads();
    for (int s = 128; s > 0; s >>= 1) { if (tid < s) red[tid] += red[tid + s]; __syncthreads(); }
    float var = red[0] / E; __syncthreads();
    float y = 0.0f;
    if (tid < E) {
        y = d * rsqrtf(var + 1e-5f) * g[tid] + b[tid];
        out[(size_t)row * E + tid] = y;
    }
    red[tid] = fabsf(y); __syncthreads();
    for (int s = 128; s > 0; s >>= 1) {
        if (tid < s) red[tid] = fmaxf(red[tid], red[tid + s]);
        __syncthreads();
    }
    if (tid == 0) atomicMax(amax, __float_as_uint(red[0]));
}

__global__ void k_quant(const float* __restrict__ x, int ldx, int rowsPad, int rowsValid,
                        int cols, int8_t* __restrict__ q, int ldq,
                        const unsigned* __restrict__ amax, float* __restrict__ s_out) {
    float s = cell_scale(amax);
    long total = (long)rowsPad * ldq;
    for (long i = blockIdx.x * (long)blockDim.x + threadIdx.x; i < total;
         i += (long)gridDim.x * blockDim.x) {
        int r = (int)(i / ldq), c = (int)(i % ldq);
        int8_t v = 0;
        if (r < rowsValid && c < cols) {
            float t = rintf(x[(size_t)r * ldx + c] / s);
            t = fminf(fmaxf(t, -128.0f), 127.0f);
            v = (int8_t)(int)t;
        }
        q[i] = v;
    }
    if (blockIdx.x == 0 && threadIdx.x == 0) s_out[0] = s;
}

template <int KSTEPS, int LDA, int LDB>
__global__ void k_gemm_i8(const int8_t* __restrict__ A,
                          const int8_t* __restrict__ Bq,
                          const float* __restrict__ sA, const float* __restrict__ sB,
                          const float* __restrict__ bias,
                          const float* __restrict__ resid,
                          float* __restrict__ out, int ldo,
                          int M, int N, int nGroups) {
    int wave = blockIdx.x * (blockDim.x >> 5) + (threadIdx.x >> 5);
    int tilesM = M >> 4;
    if (wave >= tilesM * nGroups) return;
    int tm = wave / nGroups, g = wave % nGroups;
    int lane = threadIdx.x & 31;
    int half = lane >> 4, l16 = lane & 15;

    A  = (const int8_t*)__builtin_assume_aligned(A, 256);
    Bq = (const int8_t*)__builtin_assume_aligned(Bq, 256);
    const int8_t* arow  = A  + (size_t)(tm * 16 + l16) * LDA;
    const int8_t* brow0 = Bq + (size_t)(g * 64 + l16) * LDB;

    __builtin_prefetch(arow, 0, 1);
    __builtin_prefetch(brow0, 0, 1);

    auto loadA = [&](int kb, v8i& av) {
#pragma unroll
        for (int j = 0; j < 8; ++j) {
            int ka = kb + ((j >> 1) << 4) + ((j & 1) << 2) + (half << 3);
            av[j] = *reinterpret_cast<const int*>(arow + ka);
        }
    };
    auto loadB = [&](const int8_t* br, int kb, v8i& bv) {
#pragma unroll
        for (int j = 0; j < 8; ++j) {
            int kbo = kb + ((j >> 1) << 4) + ((j & 1) << 2) + (half << 3);
            bv[j] = *reinterpret_cast<const int*>(br + kbo);
        }
    };

    v8i acc0 = {0, 0, 0, 0, 0, 0, 0, 0};
    v8i acc1 = acc0, acc2 = acc0, acc3 = acc0;

    v8i aPing[2];
    loadA(0, aPing[0]);
#pragma unroll
    for (int ks = 0; ks < KSTEPS; ++ks) {
        int kb = ks << 6;
        v8i b0, b1, b2, b3;
        loadB(brow0,            kb, b0);
        loadB(brow0 + 16 * LDB, kb, b1);
        loadB(brow0 + 32 * LDB, kb, b2);
        loadB(brow0 + 48 * LDB, kb, b3);
        if (ks + 1 < KSTEPS) loadA((ks + 1) << 6, aPing[(ks + 1) & 1]);
        const v8i& a = aPing[ks & 1];
        acc0 = __builtin_amdgcn_wmma_i32_16x16x64_iu8(true, a, true, b0, acc0, false, false);
        acc1 = __builtin_amdgcn_wmma_i32_16x16x64_iu8(true, a, true, b1, acc1, false, false);
        acc2 = __builtin_amdgcn_wmma_i32_16x16x64_iu8(true, a, true, b2, acc2, false, false);
        acc3 = __builtin_amdgcn_wmma_i32_16x16x64_iu8(true, a, true, b3, acc3, false, false);
    }

    float s = sA[0] * sB[0];
    auto emit = [&](const v8i& acc, int t) {
        int n = (g * 4 + t) * 16 + l16;
        if (n < N) {
            float bn = bias ? bias[n] : 0.0f;
#pragma unroll
            for (int r = 0; r < 8; ++r) {
                int m = tm * 16 + r + (half << 3);
                float v = s * (float)acc[r] + bn;
                size_t o = (size_t)m * ldo + n;
                if (resid) v += resid[o];
                out[o] = v;
            }
        }
    };
    emit(acc0, 0); emit(acc1, 1); emit(acc2, 2); emit(acc3, 3);
}

__global__ void k_attn(const float* __restrict__ qb, const float* __restrict__ kb,
                       const float* __restrict__ vb, float* __restrict__ ob,
                       int ld, int mode, unsigned* __restrict__ amax) {
    using namespace cfg;
    const int L = mode ? C : T;
    int seq = blockIdx.x / H, head = blockIdx.x % H;
    __shared__ float qs[14 * 36], ks2[14 * 36], vs[14 * 36], sc[14 * 14], red[512];
    int tid = threadIdx.x;
    int rowbase, rowstride;
    if (mode == 0) { int b = seq / C, c = seq % C; rowbase = b * S + c * T; rowstride = 1; }
    else           { int b = seq / T, t = seq % T; rowbase = b * S + t;     rowstride = T; }
    int col0 = head * DH;
    for (int i = tid; i < L * DH; i += blockDim.x) {
        int l = i / DH, d = i % DH;
        size_t r = (size_t)(rowbase + l * rowstride) * ld + col0 + d;
        qs[i] = qb[r]; ks2[i] = kb[r]; vs[i] = vb[r];
    }
    __syncthreads();
    for (int i = tid; i < L * L; i += blockDim.x) {
        int l = i / L, m = i % L;
        float a = 0.0f;
        for (int d = 0; d < DH; ++d) a += qs[l * DH + d] * ks2[m * DH + d];
        sc[i] = a * (1.0f / 6.0f);
    }
    __syncthreads();
    if (tid < L) {
        float mx = -1e30f;
        for (int m = 0; m < L; ++m) mx = fmaxf(mx, sc[tid * L + m]);
        float sm = 0.0f;
        for (int m = 0; m < L; ++m) { float e = __expf(sc[tid * L + m] - mx); sc[tid * L + m] = e; sm += e; }
        float inv = 1.0f / sm;
        for (int m = 0; m < L; ++m) sc[tid * L + m] *= inv;
    }
    __syncthreads();
    float lm = 0.0f;
    for (int i = tid; i < L * DH; i += blockDim.x) {
        int l = i / DH, d = i % DH;
        float a = 0.0f;
        for (int m = 0; m < L; ++m) a += sc[l * L + m] * vs[m * DH + d];
        ob[(size_t)(rowbase + l * rowstride) * ld + col0 + d] = a;
        lm = fmaxf(lm, fabsf(a));
    }
    red[tid] = lm; __syncthreads();
    for (int s = 256; s > 0; s >>= 1) {
        if (tid < s) red[tid] = fmaxf(red[tid], red[tid + s]);
        __syncthreads();
    }
    if (tid == 0) atomicMax(amax, __float_as_uint(red[0]));
}

__global__ void k_gelu(float* __restrict__ x, long n, unsigned* __restrict__ amax) {
    __shared__ float red[256];
    float m = 0.0f;
    for (long i = blockIdx.x * (long)blockDim.x + threadIdx.x; i < n;
         i += (long)gridDim.x * blockDim.x) {
        float t = x[i];
        float u = 0.7978845608028654f * (t + 0.044715f * t * t * t);
        float y = 0.5f * t * (1.0f + tanhf(u));
        x[i] = y;
        m = fmaxf(m, fabsf(y));
    }
    red[threadIdx.x] = m; __syncthreads();
    for (int s = 128; s > 0; s >>= 1) {
        if (threadIdx.x < s) red[threadIdx.x] = fmaxf(red[threadIdx.x], red[threadIdx.x + s]);
        __syncthreads();
    }
    if (threadIdx.x == 0) atomicMax(amax, __float_as_uint(red[0]));
}

__global__ void k_proj(const float* __restrict__ x, const float* __restrict__ w,
                       const float* __restrict__ bias,
                       const unsigned* amX, const unsigned* amW, float* __restrict__ h) {
    using namespace cfg;
    int idx = blockIdx.x * blockDim.x + threadIdx.x;
    if (idx >= B * T * E) return;
    int e = idx % E, t = (idx / E) % T, b = idx / (E * T);
    float sx = cell_scale(amX), sw = cell_scale(amW);
    float acc = 0.0f;
    for (int c = 0; c < C; ++c)
        acc += fakeq(x[((size_t)b * C + c) * T + t], sx) * fakeq(w[e * C + c], sw);
    acc += bias[e];
    for (int c = 0; c < C; ++c)
        h[((size_t)b * S + c * T + t) * E + e] = acc;
}

__global__ void k_pool(const float* __restrict__ a, const unsigned* amaxA,
                       float* __restrict__ pooled, unsigned* __restrict__ amaxP) {
    using namespace cfg;
    __shared__ float red[256];
    int idx = blockIdx.x * blockDim.x + threadIdx.x;
    float lm = 0.0f;
    if (idx < B * E) {
        int b = idx / E, e = idx % E;
        float s = cell_scale(amaxA);
        float acc = 0.0f;
        for (int t = 0; t < S; ++t) acc += fakeq(a[((size_t)b * S + t) * E + e], s);
        acc *= (1.0f / S);
        pooled[idx] = acc;
        lm = fabsf(acc);
    }
    red[threadIdx.x] = lm; __syncthreads();
    for (int s = 128; s > 0; s >>= 1) {
        if (threadIdx.x < s) red[threadIdx.x] = fmaxf(red[threadIdx.x], red[threadIdx.x + s]);
        __syncthreads();
    }
    if (threadIdx.x == 0) atomicMax(amaxP, __float_as_uint(red[0]));
}

__global__ void k_cls(const float* __restrict__ pooled, const float* __restrict__ w,
                      const float* __restrict__ bias,
                      const unsigned* amP, const unsigned* amW, float* __restrict__ out) {
    using namespace cfg;
    int idx = blockIdx.x * blockDim.x + threadIdx.x;
    if (idx >= B * NC) return;
    int b = idx / NC, n = idx % NC;
    float sp = cell_scale(amP), sw = cell_scale(amW);
    float acc = 0.0f;
    for (int e = 0; e < E; ++e)
        acc += fakeq(pooled[(size_t)b * E + e], sp) * fakeq(w[n * E + e], sw);
    out[idx] = acc + bias[n];
}

extern "C" void kernel_launch(void* const* d_in, const int* in_sizes, int n_in,
                              void* d_out, int out_size, void* d_ws, size_t ws_size,
                              hipStream_t stream) {
    using namespace cfg;
    (void)in_sizes; (void)n_in; (void)out_size; (void)ws_size;

    const float* x      = (const float*)d_in[0];
    const float* proj_w = (const float*)d_in[1];
    const float* proj_b = (const float*)d_in[2];
    const float* ln1_g  = (const float*)d_in[3];
    const float* ln1_b  = (const float*)d_in[4];
    const float* wq     = (const float*)d_in[5];
    const float* bq     = (const float*)d_in[6];
    const float* wk     = (const float*)d_in[7];
    const float* bk     = (const float*)d_in[8];
    const float* wv     = (const float*)d_in[9];
    const float* bv     = (const float*)d_in[10];
    const float* wo     = (const float*)d_in[11];
    const float* bo     = (const float*)d_in[12];
    const float* ln2_g  = (const float*)d_in[13];
    const float* ln2_b  = (const float*)d_in[14];
    const float* fc1_w  = (const float*)d_in[15];
    const float* fc1_b  = (const float*)d_in[16];
    const float* fc2_w  = (const float*)d_in[17];
    const float* fc2_b  = (const float*)d_in[18];
    const float* fn_g   = (const float*)d_in[19];
    const float* fn_b   = (const float*)d_in[20];
    const float* cls_w  = (const float*)d_in[21];
    const float* cls_b  = (const float*)d_in[22];
    float* out = (float*)d_out;

    uint8_t* ws = (uint8_t*)d_ws;
    size_t off = 0;
    auto carve = [&](size_t bytes) -> uint8_t* {
        off = (off + 255) & ~(size_t)255;
        uint8_t* p = ws + off;
        off += bytes;
        return p;
    };
    float*   buf_h   = (float*)carve((size_t)M * E * 4);
    float*   buf_f1  = (float*)carve((size_t)M * E * 4);
    float*   buf_big = (float*)carve((size_t)M * MLP * 4);
    int8_t*  qa      = (int8_t*)carve((size_t)M * KP_MLP);
    int8_t*  qw      = (int8_t*)carve((size_t)NP_MLP * KP_MLP);
    float*   pooled  = (float*)carve((size_t)B * E * 4);
    unsigned* cells  = (unsigned*)carve(64 * 4);
    float*   fcells  = (float*)cells;

    auto cdiv = [](long a, long b) { return (int)((a + b - 1) / b); };
    auto qgrid = [&](long n) { int g = cdiv(n, 256); return g > 32768 ? 32768 : g; };

    const int tilesM = M / 16;
    const int nGroups_E   = NP_E / 64;
    const int nGroups_MLP = NP_MLP / 64;
    const int gemmBlocks_E   = cdiv((long)tilesM * nGroups_E, 8);
    const int gemmBlocks_MLP = cdiv((long)tilesM * nGroups_MLP, 8);

    k_reset<<<1, 64, 0, stream>>>(cells, 64);
    k_absmax<<<qgrid(B * C * T), 256, 0, stream>>>(x, (long)B * C * T, cells + 10);
    k_absmax<<<qgrid(E * C), 256, 0, stream>>>(proj_w, (long)E * C, cells + 11);
    k_proj<<<cdiv((long)B * T * E, 256), 256, 0, stream>>>(x, proj_w, proj_b,
                                                           cells + 10, cells + 11, buf_h);

    for (int i = 0; i < NB; ++i) {
        const float* Wq = wq + (size_t)i * E * E;
        const float* Wk = wk + (size_t)i * E * E;
        const float* Wv = wv + (size_t)i * E * E;
        const float* Wo = wo + (size_t)i * E * E;
        const float* W1 = fc1_w + (size_t)i * MLP * E;
        const float* W2 = fc2_w + (size_t)i * E * MLP;

        k_reset<<<1, 64, 0, stream>>>(cells, 16);

        k_ln<<<M, 256, 0, stream>>>(buf_h, ln1_g + i * E, ln1_b + i * E, buf_f1, cells + 0);
        k_quant<<<qgrid((long)M * KP_E), 256, 0, stream>>>(buf_f1, E, M, M, E,
                                                           qa, KP_E, cells + 0, fcells + 16);
        k_absmax<<<qgrid(E * E), 256, 0, stream>>>(Wq, (long)E * E, cells + 1);
        k_quant<<<qgrid((long)NP_E * KP_E), 256, 0, stream>>>(Wq, E, NP_E, E, E,
                                                              qw, KP_E, cells + 1, fcells + 17);
        k_gemm_i8<3, KP_E, KP_E><<<gemmBlocks_E, 256, 0, stream>>>(
            qa, qw, fcells + 16, fcells + 17, bq + i * E, nullptr,
            buf_big + 0, MLP, M, E, nGroups_E);
        k_absmax<<<qgrid(E * E), 256, 0, stream>>>(Wk, (long)E * E, cells + 2);
        k_quant<<<qgrid((long)NP_E * KP_E), 256, 0, stream>>>(Wk, E, NP_E, E, E,
                                                              qw, KP_E, cells + 2, fcells + 18);
        k_gemm_i8<3, KP_E, KP_E><<<gemmBlocks_E, 256, 0, stream>>>(
            qa, qw, fcells + 16, fcells + 18, bk + i * E, nullptr,
            buf_big + E, MLP, M, E, nGroups_E);
        k_absmax<<<qgrid(E * E), 256, 0, stream>>>(Wv, (long)E * E, cells + 3);
        k_quant<<<qgrid((long)NP_E * KP_E), 256, 0, stream>>>(Wv, E, NP_E, E, E,
                                                              qw, KP_E, cells + 3, fcells + 19);
        k_gemm_i8<3, KP_E, KP_E><<<gemmBlocks_E, 256, 0, stream>>>(
            qa, qw, fcells + 16, fcells + 19, bv + i * E, nullptr,
            buf_big + 2 * E, MLP, M, E, nGroups_E);
        int mode = i & 1;
        int nseq = mode ? (B * T) : (B * C);
        k_attn<<<nseq * H, 512, 0, stream>>>(buf_big, buf_big + E, buf_big + 2 * E,
                                             buf_big + 3 * E, MLP, mode, cells + 4);
        k_quant<<<qgrid((long)M * KP_E), 256, 0, stream>>>(buf_big + 3 * E, MLP, M, M, E,
                                                           qa, KP_E, cells + 4, fcells + 20);
        k_absmax<<<qgrid(E * E), 256, 0, stream>>>(Wo, (long)E * E, cells + 5);
        k_quant<<<qgrid((long)NP_E * KP_E), 256, 0, stream>>>(Wo, E, NP_E, E, E,
                                                              qw, KP_E, cells + 5, fcells + 21);
        k_gemm_i8<3, KP_E, KP_E><<<gemmBlocks_E, 256, 0, stream>>>(
            qa, qw, fcells + 20, fcells + 21, bo + i * E, buf_h,
            buf_h, E, M, E, nGroups_E);

        k_ln<<<M, 256, 0, stream>>>(buf_h, ln2_g + i * E, ln2_b + i * E, buf_f1, cells + 6);
        k_quant<<<qgrid((long)M * KP_E), 256, 0, stream>>>(buf_f1, E, M, M, E,
                                                           qa, KP_E, cells + 6, fcells + 22);
        k_absmax<<<qgrid(MLP * E), 256, 0, stream>>>(W1, (long)MLP * E, cells + 7);
        k_quant<<<qgrid((long)NP_MLP * KP_E), 256, 0, stream>>>(W1, E, NP_MLP, MLP, E,
                                                                qw, KP_E, cells + 7, fcells + 23);
        k_gemm_i8<3, KP_E, KP_E><<<gemmBlocks_MLP, 256, 0, stream>>>(
            qa, qw, fcells + 22, fcells + 23, fc1_b + i * MLP, nullptr,
            buf_big, MLP, M, MLP, nGroups_MLP);
        k_gelu<<<qgrid((long)M * MLP), 256, 0, stream>>>(buf_big, (long)M * MLP, cells + 8);
        k_quant<<<qgrid((long)M * KP_MLP), 256, 0, stream>>>(buf_big, MLP, M, M, MLP,
                                                             qa, KP_MLP, cells + 8, fcells + 24);
        k_absmax<<<qgrid(E * MLP), 256, 0, stream>>>(W2, (long)E * MLP, cells + 9);
        k_quant<<<qgrid((long)NP_E * KP_MLP), 256, 0, stream>>>(W2, MLP, NP_E, E, MLP,
                                                                qw, KP_MLP, cells + 9, fcells + 25);
        k_gemm_i8<12, KP_MLP, KP_MLP><<<gemmBlocks_E, 256, 0, stream>>>(
            qa, qw, fcells + 24, fcells + 25, fc2_b + i * E, buf_h,
            buf_h, E, M, E, nGroups_E);
    }

    k_reset<<<1, 64, 0, stream>>>(cells, 16);
    k_ln<<<M, 256, 0, stream>>>(buf_h, fn_g, fn_b, buf_f1, cells + 12);
    k_pool<<<cdiv((long)B * E, 256), 256, 0, stream>>>(buf_f1, cells + 12, pooled, cells + 13);
    k_absmax<<<qgrid(NC * E), 256, 0, stream>>>(cls_w, (long)NC * E, cells + 14);
    k_cls<<<cdiv((long)B * NC, 256), 256, 0, stream>>>(pooled, cls_w, cls_b,
                                                       cells + 13, cells + 14, out);
}
